// SwinTransformerBlockDISTILL_6012954214727
// MI455X (gfx1250) — matrix operand order repaired and checked by reading, not compiled, not run on hardware
//
#include <hip/hip_runtime.h>

typedef __attribute__((ext_vector_type(16))) _Float16 v16h;
typedef __attribute__((ext_vector_type(8)))  float    v8f;

// ---------------- problem constants ----------------
constexpr int BB = 32, HH = 56, WW2 = 56, D = 192, WS = 7, NH = 6, SHIFT = 3;
constexpr int DH = D / NH;            // 32
constexpr int NT = WS * WS;           // 49 tokens per window
constexpr int NWIN_PER_B = (HH/WS)*(WW2/WS);   // 64
constexpr int NWIN = BB * NWIN_PER_B;          // 2048
constexpr int MTOK = NWIN * NT;                // 100352 rows
constexpr size_t S_ELEMS = (size_t)MTOK * D;   // 19,267,584 (== xo elems)
constexpr float QK_SCALE = 0.17677669529663687f; // 1/sqrt(32)

// ---------------- workspace layout (bytes) ----------------
constexpr size_t O_QKVT   = 0;              // 576x192 f16
constexpr size_t O_PROJT  = 221184;         // 192x192 f16
constexpr size_t O_FC1T   = 294912;         // 768x192 f16
constexpr size_t O_FC2T   = 589824;         // 192x768 f16
constexpr size_t O_HWIN   = 884736;         // MTOK x 192 f16 (post-LN1, window order)
constexpr size_t O_SCORES = 39419904;       // NWIN x 6 x 49 x 49 f32
constexpr size_t O_ATTN16 = 157433856;      // NWIN x 6 x 49 x 49 f16
constexpr size_t O_AOUT   = 216440832;      // MTOK x 192 f16 (attn output)
constexpr size_t O_XO     = 254976000;      // MTOK x 192 f32 (post first residual)
constexpr size_t O_LN2    = 332046336;      // MTOK x 192 f16
constexpr size_t O_HID    = 370581504;      // MTOK x 768 f16

// ---------------- WMMA helpers ----------------
__device__ inline v8f wmma16(v16h a, v16h b, v8f c) {
  return __builtin_amdgcn_wmma_f32_16x16x32_f16(false, a, false, b, (short)0, c,
                                                false, false);
}

// A fragment (16x32 f16, row-major src, lda), full tile (no bounds)
__device__ inline v16h load_a_f16(const _Float16* A, int lda, int row0, int kb, int lane) {
  const _Float16* p = A + (size_t)(row0 + (lane & 15)) * lda + kb + ((lane >> 4) * 16);
  v16h a;
#pragma unroll
  for (int h = 0; h < 8; ++h) { a[h] = p[h]; a[h + 8] = p[8 + h]; }
  return a;
}

// B fragment from transposed weight Bt[N][K] (f16), full tile
__device__ inline v16h load_bt_f16(const _Float16* Bt, int ldb, int col0, int kb, int lane) {
  const _Float16* p = Bt + (size_t)(col0 + (lane & 15)) * ldb + kb + ((lane >> 4) * 16);
  v16h b;
#pragma unroll
  for (int j = 0; j < 16; ++j) b[j] = p[j];
  return b;
}

// A fragment from fp32 src with row bound + scale (for Q)
__device__ inline v16h load_a_f32b(const float* A, int lda, int rows, int row0,
                                   int koff, float scale, int lane) {
  int row = row0 + (lane & 15);
  int ks  = (lane >> 4) * 16;
  v16h a;
  if (row < rows) {
    const float* p = A + (size_t)row * lda + koff + ks;
#pragma unroll
    for (int h = 0; h < 8; ++h) {
      a[h]     = (_Float16)(p[h] * scale);
      a[h + 8] = (_Float16)(p[8 + h] * scale);
    }
  } else {
#pragma unroll
    for (int h = 0; h < 16; ++h) a[h] = (_Float16)0.f;
  }
  return a;
}

// B fragment from fp32 "Bt-style" (token-major K, i.e. scores = q.kT), col bound
__device__ inline v16h load_bt_f32b(const float* Bt, int ldb, int cols, int col0,
                                    int koff, int lane) {
  int c  = col0 + (lane & 15);
  int ks = (lane >> 4) * 16;
  v16h b;
  if (c < cols) {
    const float* p = Bt + (size_t)c * ldb + koff + ks;
#pragma unroll
    for (int j = 0; j < 16; ++j) b[j] = (_Float16)p[j];
  } else {
#pragma unroll
    for (int j = 0; j < 16; ++j) b[j] = (_Float16)0.f;
  }
  return b;
}

// A fragment f16 with row + K bounds (attn matrix, K padded 49->64)
__device__ inline v16h load_a_f16b(const _Float16* A, int lda, int rows, int row0,
                                   int kb, int kmax, int lane) {
  int row = row0 + (lane & 15);
  int ks  = (lane >> 4) * 16;
  v16h a;
#pragma unroll
  for (int h = 0; h < 8; ++h) {
    int k0 = kb + ks + h, k1 = kb + ks + 8 + h;
    a[h]     = (row < rows && k0 < kmax) ? A[(size_t)row * lda + k0] : (_Float16)0.f;
    a[h + 8] = (row < rows && k1 < kmax) ? A[(size_t)row * lda + k1] : (_Float16)0.f;
  }
  return a;
}

// B fragment from row-major fp32 B[k][col] with K bound (V matrix)
__device__ inline v16h load_b_rm_f32b(const float* Bm, int ldb, int kmax, int kb,
                                      int coloff, int lane) {
  int c  = coloff + (lane & 15);
  int ks = (lane >> 4) * 16;
  v16h b;
#pragma unroll
  for (int j = 0; j < 16; ++j) {
    int k = kb + ks + j;
    b[j] = (k < kmax) ? (_Float16)Bm[(size_t)k * ldb + c] : (_Float16)0.f;
  }
  return b;
}

// 32x64 per-wave GEMM core: A row-major f16 (lda=K), Bt f16 [N][K].
// 8 independent WMMA chains per K-step, 6 fragment loads -> 0.75 loads/WMMA.
__device__ inline void gemm_core(const _Float16* A, const _Float16* Bt, int K,
                                 int row0, int col0, int lane, v8f acc[2][4]) {
  for (int kb = 0; kb < K; kb += 32) {
    v16h a0 = load_a_f16(A, K, row0, kb, lane);
    v16h a1 = load_a_f16(A, K, row0 + 16, kb, lane);
    v16h b0 = load_bt_f16(Bt, K, col0, kb, lane);
    v16h b1 = load_bt_f16(Bt, K, col0 + 16, kb, lane);
    v16h b2 = load_bt_f16(Bt, K, col0 + 32, kb, lane);
    v16h b3 = load_bt_f16(Bt, K, col0 + 48, kb, lane);
    acc[0][0] = wmma16(a0, b0, acc[0][0]);
    acc[0][1] = wmma16(a0, b1, acc[0][1]);
    acc[0][2] = wmma16(a0, b2, acc[0][2]);
    acc[0][3] = wmma16(a0, b3, acc[0][3]);
    acc[1][0] = wmma16(a1, b0, acc[1][0]);
    acc[1][1] = wmma16(a1, b1, acc[1][1]);
    acc[1][2] = wmma16(a1, b2, acc[1][2]);
    acc[1][3] = wmma16(a1, b3, acc[1][3]);
  }
}

// token (window order) -> flat image row index (b*3136 + h*56 + w).
// roll(-3) gather for LN1 and roll(+3) scatter for window-reverse are the same map.
__device__ inline size_t token_img_row(int t) {
  int w    = t / NT, n = t - w * NT;
  int b    = w >> 6, widx = w & 63;
  int hwin = widx >> 3, wwin = widx & 7;
  int i = n / WS, j = n - i * WS;
  int hh = hwin * WS + i + SHIFT; if (hh >= HH)  hh -= HH;
  int ww = wwin * WS + j + SHIFT; if (ww >= WW2) ww -= WW2;
  return (size_t)b * (HH * WW2) + hh * WW2 + ww;
}

__device__ inline int regid(int p) { return p < (HH - WS) ? 0 : (p < (HH - SHIFT) ? 1 : 2); }

// ---------------- kernels ----------------
__global__ void k_transpose_w(const float* __restrict__ src, _Float16* __restrict__ dst,
                              int K, int Nc) {
  int i = blockIdx.x * 256 + threadIdx.x;
  if (i < K * Nc) {
    int k = i / Nc, n = i - k * Nc;
    dst[(size_t)n * K + k] = (_Float16)src[i];
  }
}

// LN1 + roll(-3) + window partition -> f16 (one wave per token)
__global__ void k_ln1(const float* __restrict__ x, const float* __restrict__ g,
                      const float* __restrict__ bta, _Float16* __restrict__ hwin) {
  int wave = (int)((blockIdx.x * (size_t)blockDim.x + threadIdx.x) >> 5);
  int lane = threadIdx.x & 31;
  if (wave >= MTOK) return;
  const float* xp = x + token_img_row(wave) * D;
  float v[6], s = 0.f;
#pragma unroll
  for (int i = 0; i < 6; ++i) { v[i] = xp[lane + 32 * i]; s += v[i]; }
#pragma unroll
  for (int off = 16; off >= 1; off >>= 1) s += __shfl_xor(s, off, 32);
  float mu = s * (1.f / D);
  float var = 0.f;
#pragma unroll
  for (int i = 0; i < 6; ++i) { float d = v[i] - mu; var += d * d; }
#pragma unroll
  for (int off = 16; off >= 1; off >>= 1) var += __shfl_xor(var, off, 32);
  float rs = rsqrtf(var * (1.f / D) + 1e-5f);
#pragma unroll
  for (int i = 0; i < 6; ++i) {
    int c = lane + 32 * i;
    hwin[(size_t)wave * D + c] = (_Float16)((v[i] - mu) * rs * g[c] + bta[c]);
  }
}

// LN2 over xo rows (identity mapping)
__global__ void k_ln2(const float* __restrict__ xo, const float* __restrict__ g,
                      const float* __restrict__ bta, _Float16* __restrict__ dst) {
  int wave = (int)((blockIdx.x * (size_t)blockDim.x + threadIdx.x) >> 5);
  int lane = threadIdx.x & 31;
  if (wave >= MTOK) return;
  const float* xp = xo + (size_t)wave * D;
  float v[6], s = 0.f;
#pragma unroll
  for (int i = 0; i < 6; ++i) { v[i] = xp[lane + 32 * i]; s += v[i]; }
#pragma unroll
  for (int off = 16; off >= 1; off >>= 1) s += __shfl_xor(s, off, 32);
  float mu = s * (1.f / D);
  float var = 0.f;
#pragma unroll
  for (int i = 0; i < 6; ++i) { float d = v[i] - mu; var += d * d; }
#pragma unroll
  for (int off = 16; off >= 1; off >>= 1) var += __shfl_xor(var, off, 32);
  float rs = rsqrtf(var * (1.f / D) + 1e-5f);
#pragma unroll
  for (int i = 0; i < 6; ++i) {
    int c = lane + 32 * i;
    dst[(size_t)wave * D + c] = (_Float16)((v[i] - mu) * rs * g[c] + bta[c]);
  }
}

// QKV GEMM: [MTOK,192] x [192,576] -> q/k/v fp32 regions of d_out
// block: 8 waves stacked in M (256 rows), 64-col tile (never straddles q/k/v).
__global__ void k_gemm_qkv(const _Float16* __restrict__ A, const _Float16* __restrict__ Bt,
                           const float* __restrict__ bias, float* __restrict__ qbase) {
  int lane = threadIdx.x & 31, wv = threadIdx.x >> 5;
  int row0 = blockIdx.x * 256 + wv * 32;
  int col0 = blockIdx.y * 64;
  v8f acc[2][4] = {};
  gemm_core(A, Bt, D, row0, col0, lane, acc);
  int cL = lane & 15, rB = (lane >> 4) * 8;
#pragma unroll
  for (int it = 0; it < 2; ++it)
#pragma unroll
    for (int jt = 0; jt < 4; ++jt) {
      int col = col0 + jt * 16 + cL;
      int which = col / D, lc = col - which * D;
      float bv = bias[col];
      float* base = qbase + (size_t)which * S_ELEMS;
#pragma unroll
      for (int r = 0; r < 8; ++r) {
        int row = row0 + it * 16 + rB + r;
        base[(size_t)row * D + lc] = acc[it][jt][r] + bv;
      }
    }
}

// Scores: per window, q.kT * scale -> scores[win][head][49][49]
__global__ void k_scores(const float* __restrict__ q, const float* __restrict__ k,
                         float* __restrict__ scores) {
  int w = blockIdx.x;
  int wv = threadIdx.x >> 5, lane = threadIdx.x & 31;
  const float* qw = q + (size_t)w * NT * D;
  const float* kw = k + (size_t)w * NT * D;
  float* sw = scores + (size_t)w * NH * NT * NT;
  for (int idx = wv; idx < NH * 16; idx += 8) {
    int hd = idx >> 4, rem = idx & 15, mt = rem >> 2, nt = rem & 3;
    v16h a = load_a_f32b(qw, D, NT, mt * 16, hd * DH, QK_SCALE, lane);
    v16h b = load_bt_f32b(kw, D, NT, nt * 16, hd * DH, lane);
    v8f acc = {};
    acc = wmma16(a, b, acc);
    int cL = lane & 15, rB = (lane >> 4) * 8;
    int m = nt * 16 + cL;
    if (m < NT) {
#pragma unroll
      for (int r = 0; r < 8; ++r) {
        int n = mt * 16 + rB + r;
        if (n < NT) sw[(size_t)hd * NT * NT + n * NT + m] = acc[r];
      }
    }
  }
}

// +rel-pos bias, pl mix, shift mask, softmax, pw mix -> attn f16
__global__ void k_softmax(const float* __restrict__ scores, const float* __restrict__ rpb,
                          const float* __restrict__ plw, const float* __restrict__ plb,
                          const float* __restrict__ pww, const float* __restrict__ pwb,
                          _Float16* __restrict__ attn16) {
  int gid = blockIdx.x;                 // w*49 + n
  int w = gid / NT, n = gid - w * NT;
  int widx = w & 63;
  int tid = threadIdx.x;                // 64 threads
  __shared__ float sin[NH][52], smix[NH][52], smx[NH], sinv[NH];
  const float* sw = scores + ((size_t)w * NH) * (NT * NT) + n * NT;
  int ni = n / WS, nj = n - ni * WS;
  int reg_n = regid((widx >> 3) * WS + ni) * 3 + regid((widx & 7) * WS + nj);
  for (int i = tid; i < NH * NT; i += 64) {
    int h = i / NT, m = i - h * NT;
    int mi = m / WS, mj = m - mi * WS;
    int rel = (ni - mi + WS - 1) * (2 * WS - 1) + (nj - mj + WS - 1);
    sin[h][m] = sw[(size_t)h * NT * NT + m] + rpb[rel * NH + h];
  }
  __syncthreads();
  for (int i = tid; i < NH * NT; i += 64) {
    int g = i / NT, m = i - g * NT;
    float acc = plb[g];
#pragma unroll
    for (int h = 0; h < NH; ++h) acc += sin[h][m] * plw[h * NH + g];
    int mi = m / WS, mj = m - mi * WS;
    int reg_m = regid((widx >> 3) * WS + mi) * 3 + regid((widx & 7) * WS + mj);
    if (reg_m != reg_n) acc -= 100.0f;
    smix[g][m] = acc;
  }
  __syncthreads();
  if (tid < NH) {
    float mx = -1e30f;
    for (int m = 0; m < NT; ++m) mx = fmaxf(mx, smix[tid][m]);
    float s = 0.f;
    for (int m = 0; m < NT; ++m) s += expf(smix[tid][m] - mx);
    smx[tid] = mx; sinv[tid] = 1.f / s;
  }
  __syncthreads();
  for (int i = tid; i < NH * NT; i += 64) {
    int g = i / NT, m = i - g * NT;
    smix[g][m] = expf(smix[g][m] - smx[g]) * sinv[g];
  }
  __syncthreads();
  _Float16* ow = attn16 + ((size_t)w * NH) * (NT * NT) + n * NT;
  for (int i = tid; i < NH * NT; i += 64) {
    int g = i / NT, m = i - g * NT;
    float acc = pwb[g];
#pragma unroll
    for (int h = 0; h < NH; ++h) acc += smix[h][m] * pww[h * NH + g];
    ow[(size_t)g * NT * NT + m] = (_Float16)acc;
  }
}

// attn @ v -> aout f16 [MTOK][192]
__global__ void k_attnv(const _Float16* __restrict__ attn, const float* __restrict__ v,
                        _Float16* __restrict__ aout) {
  int w = blockIdx.x;
  int wv = threadIdx.x >> 5, lane = threadIdx.x & 31;
  const _Float16* aw = attn + ((size_t)w * NH) * (NT * NT);
  const float* vw = v + (size_t)w * NT * D;
  _Float16* ow = aout + (size_t)w * NT * D;
  for (int idx = wv; idx < NH * 8; idx += 8) {
    int hd = idx >> 3, rem = idx & 7, mt = rem >> 1, nt = rem & 1;
    v8f acc = {};
    for (int kb = 0; kb < 64; kb += 32) {
      v16h a = load_a_f16b(aw + (size_t)hd * NT * NT, NT, NT, mt * 16, kb, NT, lane);
      v16h b = load_b_rm_f32b(vw, D, NT, kb, hd * DH + nt * 16, lane);
      acc = wmma16(a, b, acc);
    }
    int cL = lane & 15, rB = (lane >> 4) * 8;
#pragma unroll
    for (int r = 0; r < 8; ++r) {
      int n = mt * 16 + rB + r;
      if (n < NT)
        ow[(size_t)n * D + hd * DH + nt * 16 + cL] = (_Float16)acc[r];
    }
  }
}

// proj GEMM + window reverse + roll(+3) + residual -> xo f32 (image order)
__global__ void k_gemm_proj(const _Float16* __restrict__ A, const _Float16* __restrict__ Bt,
                            const float* __restrict__ bias, const float* __restrict__ x,
                            float* __restrict__ xo) {
  int lane = threadIdx.x & 31, wv = threadIdx.x >> 5;
  int row0 = blockIdx.x * 256 + wv * 32;
  int col0 = blockIdx.y * 64;
  v8f acc[2][4] = {};
  gemm_core(A, Bt, D, row0, col0, lane, acc);
  int cL = lane & 15, rB = (lane >> 4) * 8;
#pragma unroll
  for (int it = 0; it < 2; ++it)
#pragma unroll
    for (int jt = 0; jt < 4; ++jt) {
      int col = col0 + jt * 16 + cL;
      float bv = bias[col];
#pragma unroll
      for (int r = 0; r < 8; ++r) {
        int t = row0 + it * 16 + rB + r;
        size_t dst = token_img_row(t) * D + col;
        xo[dst] = x[dst] + acc[it][jt][r] + bv;
      }
    }
}

// fc1 GEMM + exact GELU -> hidden f16 [MTOK][768]
__global__ void k_gemm_fc1(const _Float16* __restrict__ A, const _Float16* __restrict__ Bt,
                           const float* __restrict__ bias, _Float16* __restrict__ hid) {
  int lane = threadIdx.x & 31, wv = threadIdx.x >> 5;
  int row0 = blockIdx.x * 256 + wv * 32;
  int col0 = blockIdx.y * 64;
  v8f acc[2][4] = {};
  gemm_core(A, Bt, D, row0, col0, lane, acc);
  int cL = lane & 15, rB = (lane >> 4) * 8;
#pragma unroll
  for (int it = 0; it < 2; ++it)
#pragma unroll
    for (int jt = 0; jt < 4; ++jt) {
      int col = col0 + jt * 16 + cL;
      float bv = bias[col];
#pragma unroll
      for (int r = 0; r < 8; ++r) {
        int row = row0 + it * 16 + rB + r;
        float vv = acc[it][jt][r] + bv;
        float ge = 0.5f * vv * (1.0f + erff(vv * 0.7071067811865476f));
        hid[(size_t)row * (4 * D) + col] = (_Float16)ge;
      }
    }
}

// fc2 GEMM + residual -> d_out xo region (fp32)
__global__ void k_gemm_fc2(const _Float16* __restrict__ A, const _Float16* __restrict__ Bt,
                           const float* __restrict__ bias, const float* __restrict__ xo,
                           float* __restrict__ out) {
  int lane = threadIdx.x & 31, wv = threadIdx.x >> 5;
  int row0 = blockIdx.x * 256 + wv * 32;
  int col0 = blockIdx.y * 64;
  v8f acc[2][4] = {};
  gemm_core(A, Bt, 4 * D, row0, col0, lane, acc);
  int cL = lane & 15, rB = (lane >> 4) * 8;
#pragma unroll
  for (int it = 0; it < 2; ++it)
#pragma unroll
    for (int jt = 0; jt < 4; ++jt) {
      int col = col0 + jt * 16 + cL;
      float bv = bias[col];
#pragma unroll
      for (int r = 0; r < 8; ++r) {
        int row = row0 + it * 16 + rB + r;
        size_t dst = (size_t)row * D + col;
        out[dst] = xo[dst] + acc[it][jt][r] + bv;
      }
    }
}

// ---------------- launcher ----------------
extern "C" void kernel_launch(void* const* d_in, const int* in_sizes, int n_in,
                              void* d_out, int out_size, void* d_ws, size_t ws_size,
                              hipStream_t stream) {
  (void)in_sizes; (void)n_in; (void)out_size; (void)ws_size;
  const float* x       = (const float*)d_in[0];
  const float* norm1_g = (const float*)d_in[1];
  const float* norm1_b = (const float*)d_in[2];
  const float* qkv_w   = (const float*)d_in[3];
  const float* qkv_b   = (const float*)d_in[4];
  const float* proj_wo = (const float*)d_in[5];
  const float* proj_bo = (const float*)d_in[6];
  const float* rpb     = (const float*)d_in[7];
  const float* pl_w    = (const float*)d_in[8];
  const float* pl_b    = (const float*)d_in[9];
  const float* pw_w    = (const float*)d_in[10];
  const float* pw_b    = (const float*)d_in[11];
  const float* norm2_g = (const float*)d_in[12];
  const float* norm2_b = (const float*)d_in[13];
  const float* fc1_w   = (const float*)d_in[14];
  const float* fc1_b   = (const float*)d_in[15];
  const float* fc2_w   = (const float*)d_in[16];
  const float* fc2_b   = (const float*)d_in[17];

  char* ws = (char*)d_ws;
  _Float16* qkvT   = (_Float16*)(ws + O_QKVT);
  _Float16* projT  = (_Float16*)(ws + O_PROJT);
  _Float16* fc1T   = (_Float16*)(ws + O_FC1T);
  _Float16* fc2T   = (_Float16*)(ws + O_FC2T);
  _Float16* hwin   = (_Float16*)(ws + O_HWIN);
  float*    scores = (float*)   (ws + O_SCORES);
  _Float16* attn16 = (_Float16*)(ws + O_ATTN16);
  _Float16* aout   = (_Float16*)(ws + O_AOUT);
  float*    xo     = (float*)   (ws + O_XO);
  _Float16* ln2h   = (_Float16*)(ws + O_LN2);
  _Float16* hid    = (_Float16*)(ws + O_HID);

  float* outx  = (float*)d_out;
  float* qbase = outx + S_ELEMS;            // q, then k, then v (each S_ELEMS)

  // 1) weight transposes -> f16 [N][K]
  k_transpose_w<<<(192 * 576 + 255) / 256, 256, 0, stream>>>(qkv_w, qkvT, 192, 576);
  k_transpose_w<<<(192 * 192 + 255) / 256, 256, 0, stream>>>(proj_wo, projT, 192, 192);
  k_transpose_w<<<(192 * 768 + 255) / 256, 256, 0, stream>>>(fc1_w, fc1T, 192, 768);
  k_transpose_w<<<(768 * 192 + 255) / 256, 256, 0, stream>>>(fc2_w, fc2T, 768, 192);

  // 2) LN1 + roll + window partition
  k_ln1<<<MTOK / 8, 256, 0, stream>>>(x, norm1_g, norm1_b, hwin);

  // 3) QKV GEMM -> q/k/v regions of d_out
  k_gemm_qkv<<<dim3(MTOK / 256, 576 / 64), 256, 0, stream>>>(hwin, qkvT, qkv_b, qbase);

  // 4) attention scores
  k_scores<<<NWIN, 256, 0, stream>>>(qbase, qbase + S_ELEMS, scores);

  // 5) bias + pl-mix + mask + softmax + pw-mix
  k_softmax<<<MTOK, 64, 0, stream>>>(scores, rpb, pl_w, pl_b, pw_w, pw_b, attn16);

  // 6) attn @ v
  k_attnv<<<NWIN, 256, 0, stream>>>(attn16, qbase + 2 * S_ELEMS, aout);

  // 7) proj + window reverse + roll + residual -> xo
  k_gemm_proj<<<dim3(MTOK / 256, 192 / 64), 256, 0, stream>>>(aout, projT, proj_bo, x, xo);

  // 8) LN2
  k_ln2<<<MTOK / 8, 256, 0, stream>>>(xo, norm2_g, norm2_b, ln2h);

  // 9) fc1 + gelu
  k_gemm_fc1<<<dim3(MTOK / 256, 768 / 64), 256, 0, stream>>>(ln2h, fc1T, fc1_b, hid);

  // 10) fc2 + residual -> d_out
  k_gemm_fc2<<<dim3(MTOK / 256, 192 / 64), 256, 0, stream>>>(hid, fc2T, fc2_b, xo, outx);
}
